// SegmentMemoryRetriever_91139206021430
// MI455X (gfx1250) — compile-verified, matrix operand order checked by reading, not run on hardware
//
#include <hip/hip_runtime.h>
#include <hip/hip_bf16.h>
#include <math.h>

// ---------------------------------------------------------------------------
// SegmentMemoryRetriever for MI455X (gfx1250, wave32, WMMA bf16, async->LDS)
//
// Algebraically reordered: keys/values projections collapse through their
// contractions (O(B*S*H^2) -> O(B*H^2)). Remaining work:
//   - 6 bf16 WMMA GEMMs on [4096,1024] activations (~69 GFLOP)
//   - one 1 GB streaming pass over memory_bank+context_bank (~46us @ 23.3TB/s)
// GEMM tiles are moved global->LDS with GLOBAL_LOAD_ASYNC_TO_LDS_B128
// (ASYNCcnt), double-buffered; operands are pre-converted to bf16 so the
// inner loop is pure async-copy + ds_load_b128 + v_wmma.
// ---------------------------------------------------------------------------

typedef __attribute__((ext_vector_type(16))) __bf16 v16bf;
typedef __attribute__((ext_vector_type(8)))  __bf16 v8bf;
typedef __attribute__((ext_vector_type(4)))  __bf16 v4bf;
typedef __attribute__((ext_vector_type(8)))  float  v8f;
typedef __attribute__((ext_vector_type(4)))  int    v4i;

#define BH   (4096)
#define HH   (1024)
#define SS   (32)
#define LDSP (40)   // padded LDS row stride (halves) for a 32-half row

#define AS1 __attribute__((address_space(1)))
#define AS3 __attribute__((address_space(3)))

__device__ __forceinline__ v4bf cvt4_bf16(float4 v) {
    v4bf r;
    r[0] = (__bf16)v.x; r[1] = (__bf16)v.y; r[2] = (__bf16)v.z; r[3] = (__bf16)v.w;
    return r;
}

// 16B async copy global -> LDS (ASYNCcnt tracked). Builtin signature is
// (int4 AS1*, int4 AS3*, imm offset, imm cpol); cast via uintptr to enter
// the explicit address spaces from generic pointers.
__device__ __forceinline__ void async_copy16(void* lds_dst, const void* gsrc) {
#if __has_builtin(__builtin_amdgcn_global_load_async_to_lds_b128)
    __builtin_amdgcn_global_load_async_to_lds_b128(
        (AS1 v4i*)(uintptr_t)gsrc,
        (AS3 v4i*)(uint32_t)(uintptr_t)lds_dst, 0, 0);
#else
    asm volatile("global_load_async_to_lds_b128 %0, %1, off"
                 :: "v"((uint32_t)(uintptr_t)lds_dst), "v"(gsrc)
                 : "memory");
#endif
}

__device__ __forceinline__ void wait_async0() {
#if __has_builtin(__builtin_amdgcn_s_wait_asynccnt)
    __builtin_amdgcn_s_wait_asynccnt(0);
#else
    asm volatile("s_wait_asynccnt 0" ::: "memory");
#endif
}

// ---------------------------------------------------------------------------
// WMMA GEMM:  C[M,1024] = act( A[M,K] @ W[1024,K]^T + bias ),  all bf16 in.
//   A is a virtual concat of up to three [M,1024] bf16 matrices along K
//   (region chosen per 32-wide K step; for K=1024 only A0 is used).
//   Block tile 128x64, 8 waves (4x2), wave tile 32x32 = 2x2 WMMA frags.
//   Double-buffered LDS stages filled by async b128 copies.
// ACT:  0=none 1=+bias 2=relu(x+bias) 3=sigmoid(x+bias)
// OUTM: bit0 -> Cf (f32), bit1 -> Cb (bf16), bit2 -> Dd = bf16(Qd - result)
// ---------------------------------------------------------------------------
template <int K, int ACT, int OUTM>
__global__ __launch_bounds__(256)
void wmma_gemm_async(const __bf16* __restrict__ A0, const __bf16* __restrict__ A1,
                     const __bf16* __restrict__ A2, const __bf16* __restrict__ W,
                     const float* __restrict__ bias,
                     float* __restrict__ Cf, __bf16* __restrict__ Cb,
                     const float* __restrict__ Qd, __bf16* __restrict__ Dd)
{
    constexpr int N = HH;
    constexpr int KSTEPS = K / 32;

    __shared__ alignas(16) __bf16 sA[2][128 * LDSP];
    __shared__ alignas(16) __bf16 sB[2][ 64 * LDSP];

    const int tid   = threadIdx.x;
    const int lane  = tid & 31;
    const int wid   = tid >> 5;
    const int waveM = wid >> 1;
    const int waveN = wid &  1;
    const int mBlk  = blockIdx.x * 128;
    const int nBlk  = blockIdx.y * 64;

    const int l15 = lane & 15;
    const int khi = (lane >> 4) * 8;   // lanes 16-31: K halves {8..15,24..31}

    v8f acc[2][2] = {};

    // stage issue: A tile 128x32 bf16 (8KB -> 2 chunks/thr), B tile 64x32 (1 chunk/thr)
    auto issue = [&](int kt, int buf) {
        const int kk0 = kt * 32;
        const int region = kk0 >> 10;                 // which concat matrix
        const __bf16* Ab = (region == 0) ? A0 : (region == 1) ? A1 : A2;
        const int kin = kk0 & (HH - 1);
        #pragma unroll
        for (int i = 0; i < 2; ++i) {
            int c   = tid + (i << 8);
            int row = c >> 2;
            int ko  = (c & 3) * 8;
            async_copy16(&sA[buf][row * LDSP + ko],
                         Ab + (size_t)(mBlk + row) * HH + kin + ko);
        }
        {
            int row = tid >> 2;
            int ko  = (tid & 3) * 8;
            async_copy16(&sB[buf][row * LDSP + ko],
                         W + (size_t)(nBlk + row) * K + kk0 + ko);
        }
    };

    issue(0, 0);
    wait_async0();
    __syncthreads();

    for (int kt = 0; kt < KSTEPS; ++kt) {
        const int cur = kt & 1;
        if (kt + 1 < KSTEPS) issue(kt + 1, cur ^ 1);

        v16bf aF[2], bF[2];
        #pragma unroll
        for (int i = 0; i < 2; ++i) {
            const __bf16* pa = &sA[cur][(waveM * 32 + i * 16 + l15) * LDSP + khi];
            v8bf lo = *(const v8bf*)(pa);
            v8bf hi = *(const v8bf*)(pa + 16);
            aF[i] = __builtin_shufflevector(lo, hi, 0,1,2,3,4,5,6,7,8,9,10,11,12,13,14,15);
        }
        #pragma unroll
        for (int j = 0; j < 2; ++j) {
            const __bf16* pb = &sB[cur][(waveN * 32 + j * 16 + l15) * LDSP + khi];
            v8bf lo = *(const v8bf*)(pb);
            v8bf hi = *(const v8bf*)(pb + 16);
            bF[j] = __builtin_shufflevector(lo, hi, 0,1,2,3,4,5,6,7,8,9,10,11,12,13,14,15);
        }

        #pragma unroll
        for (int i = 0; i < 2; ++i)
            #pragma unroll
            for (int j = 0; j < 2; ++j)
                acc[i][j] = __builtin_amdgcn_wmma_f32_16x16x32_bf16(
                    false, aF[i], false, bF[j], (short)0, acc[i][j], false, false);

        wait_async0();     // own async copies for next stage landed
        __syncthreads();   // whole workgroup done writing/reading stages
    }

    // epilogue: D frag -> lanes0-15 M=r, lanes16-31 M=r+8, N=lane%16
    const int rowHalf = (lane >> 4) * 8;
    #pragma unroll
    for (int j = 0; j < 2; ++j) {
        int col  = nBlk + waveN * 32 + j * 16 + l15;
        float bv = (ACT >= 1) ? bias[col] : 0.0f;
        #pragma unroll
        for (int i = 0; i < 2; ++i) {
            #pragma unroll
            for (int r = 0; r < 8; ++r) {
                int rowg   = mBlk + waveM * 32 + i * 16 + rowHalf + r;
                size_t idx = (size_t)rowg * N + col;
                float x = acc[i][j][r];
                if (ACT == 1) x += bv;
                if (ACT == 2) x = fmaxf(x + bv, 0.0f);
                if (ACT == 3) { x += bv; x = 1.0f / (1.0f + __expf(-x)); }
                if constexpr (OUTM & 1) Cf[idx] = x;
                if constexpr (OUTM & 2) Cb[idx] = (__bf16)x;
                if constexpr (OUTM & 4) Dd[idx] = (__bf16)(Qd[idx] - x);
            }
        }
    }
}

// ---------------------------------------------------------------------------
// Streaming pass (bandwidth-bound, 1 GB):  per batch row b:
//   scores[s] = (qk_b . context[b,s,:]) / 32 ; attn = softmax -> output
//   pooled_bf16[b] = sum_s attn[s] * memory_bank[b,s,:]
// ---------------------------------------------------------------------------
__global__ __launch_bounds__(256)
void attn_pool(const float* __restrict__ qk, const float* __restrict__ ctx,
               const float* __restrict__ mem, float* __restrict__ attn_out,
               __bf16* __restrict__ pooled)
{
    const int b    = blockIdx.x;
    const int tid  = threadIdx.x;
    const int lane = tid & 31;
    const int wid  = tid >> 5;

    __shared__ float s_sc[SS];
    __shared__ float s_at[SS];

    const float4* q4 = (const float4*)(qk + (size_t)b * HH);

    #pragma unroll
    for (int ss = 0; ss < 4; ++ss) {
        int s = wid * 4 + ss;
        const float4* c4 = (const float4*)(ctx + ((size_t)b * SS + s) * HH);
        float acc = 0.0f;
        #pragma unroll
        for (int j = 0; j < 8; ++j) {
            float4 a = q4[lane + j * 32];
            float4 c = c4[lane + j * 32];
            acc += a.x * c.x + a.y * c.y + a.z * c.z + a.w * c.w;
        }
        #pragma unroll
        for (int m = 16; m; m >>= 1) acc += __shfl_xor(acc, m, 32);
        if (lane == 0) s_sc[s] = acc * 0.03125f;   // 1/sqrt(1024)
    }
    __syncthreads();

    if (tid < SS) {
        float mx = -1e30f;
        #pragma unroll
        for (int s = 0; s < SS; ++s) mx = fmaxf(mx, s_sc[s]);
        float sum = 0.0f;
        #pragma unroll
        for (int s = 0; s < SS; ++s) sum += __expf(s_sc[s] - mx);
        float a = __expf(s_sc[tid] - mx) / sum;
        s_at[tid] = a;
        attn_out[(size_t)b * SS + tid] = a;
    }
    __syncthreads();

    const float4* m4 = (const float4*)(mem + (size_t)b * SS * HH);
    float4 p = make_float4(0.f, 0.f, 0.f, 0.f);
    #pragma unroll 8
    for (int s = 0; s < SS; ++s) {
        float w  = s_at[s];
        float4 v = m4[s * (HH / 4) + tid];
        p.x += w * v.x; p.y += w * v.y; p.z += w * v.z; p.w += w * v.w;
    }
    ((v4bf*)(pooled + (size_t)b * HH))[tid] = cvt4_bf16(p);
}

// f32 -> bf16 bulk convert
__global__ __launch_bounds__(256)
void cvt_bf16_kernel(const float* __restrict__ in, __bf16* __restrict__ out, int n4)
{
    int i = blockIdx.x * blockDim.x + threadIdx.x;
    if (i >= n4) return;
    ((v4bf*)out)[i] = cvt4_bf16(((const float4*)in)[i]);
}

// 1024x1024: Wk (f32) -> WkT (bf16), so qk = q @ WkT^T fits the uniform GEMM
__global__ __launch_bounds__(256)
void transpose1024_bf(const float* __restrict__ in, __bf16* __restrict__ out)
{
    __shared__ float tile[32][33];
    int x = blockIdx.x * 32 + threadIdx.x;
    int y = blockIdx.y * 32 + threadIdx.y;
    #pragma unroll
    for (int j = 0; j < 32; j += 8)
        tile[threadIdx.y + j][threadIdx.x] = in[(size_t)(y + j) * HH + x];
    __syncthreads();
    x = blockIdx.y * 32 + threadIdx.x;
    y = blockIdx.x * 32 + threadIdx.y;
    #pragma unroll
    for (int j = 0; j < 32; j += 8)
        out[(size_t)(y + j) * HH + x] = (__bf16)tile[threadIdx.x][threadIdx.y + j];
}

// fused = gate*query + (1-gate)*retrieved   (bf16 out; feeds final GEMM)
__global__ __launch_bounds__(256)
void fuse_gate(const float* __restrict__ g, const float* __restrict__ q,
               const float* __restrict__ r, __bf16* __restrict__ out, int n4)
{
    int i = blockIdx.x * blockDim.x + threadIdx.x;
    if (i >= n4) return;
    float4 gg = ((const float4*)g)[i];
    float4 qq = ((const float4*)q)[i];
    float4 rr = ((const float4*)r)[i];
    float4 o;
    o.x = gg.x * qq.x + (1.0f - gg.x) * rr.x;
    o.y = gg.y * qq.y + (1.0f - gg.y) * rr.y;
    o.z = gg.z * qq.z + (1.0f - gg.z) * rr.z;
    o.w = gg.w * qq.w + (1.0f - gg.w) * rr.w;
    ((v4bf*)out)[i] = cvt4_bf16(o);
}

// ---------------------------------------------------------------------------
extern "C" void kernel_launch(void* const* d_in, const int* in_sizes, int n_in,
                              void* d_out, int out_size, void* d_ws, size_t ws_size,
                              hipStream_t stream)
{
    const float* query        = (const float*)d_in[0];
    const float* memory_bank  = (const float*)d_in[1];
    const float* context_bank = (const float*)d_in[2];
    const float* Wq  = (const float*)d_in[3];
    const float* Wk  = (const float*)d_in[4];
    const float* Wv  = (const float*)d_in[5];
    const float* Wg1 = (const float*)d_in[6];   // [1024, 3072] row-major
    const float* bg1 = (const float*)d_in[7];
    const float* Wg2 = (const float*)d_in[8];
    const float* bg2 = (const float*)d_in[9];
    const float* Wo  = (const float*)d_in[10];
    const float* bo  = (const float*)d_in[11];

    float* out  = (float*)d_out;                 // [4096,1024]
    float* attn = out + (size_t)BH * HH;         // [4096,32]

    // ---- workspace bump allocator (256B aligned slots) ----
    char* wsp = (char*)d_ws;
    auto alloc = [&](size_t bytes) -> char* {
        char* p = wsp; wsp += (bytes + 255) & ~(size_t)255; return p;
    };
    const size_t AV = (size_t)BH * HH;           // activation element count
    __bf16* qbf      = (__bf16*)alloc(AV * 2);   // bf16(query)
    __bf16* qpbf     = (__bf16*)alloc(AV * 2);   // bf16(q = query@Wq^T)  -> reused: fusedbf
    float*  qk       = (float* )alloc(AV * 4);   // f32, softmax input    -> reused: gate
    __bf16* pooledbf = (__bf16*)alloc(AV * 2);
    float*  retrf    = (float* )alloc(AV * 4);
    __bf16* retrbf   = (__bf16*)alloc(AV * 2);
    __bf16* diffbf   = (__bf16*)alloc(AV * 2);   // bf16(query - retrieved)
    __bf16* g1bf     = (__bf16*)alloc(AV * 2);
    __bf16* Wqb      = (__bf16*)alloc((size_t)HH * HH * 2);
    __bf16* WkTb     = (__bf16*)alloc((size_t)HH * HH * 2);
    __bf16* Wvb      = (__bf16*)alloc((size_t)HH * HH * 2);
    __bf16* Wg1b     = (__bf16*)alloc((size_t)HH * 3 * HH * 2);
    __bf16* Wg2b     = (__bf16*)alloc((size_t)HH * HH * 2);
    __bf16* Wob      = (__bf16*)alloc((size_t)HH * HH * 2);
    __bf16* fusedbf  = qpbf;                     // q dead after qk GEMM
    float*  gate     = qk;                       // qk dead after attn_pool

    const int W4  = HH * HH / 4;                 // float4 count of a 1024x1024 weight
    const int A4  = BH * HH / 4;
    dim3 gemmGrid(BH / 128, HH / 64);            // 32 x 16

    // ---- one-time bf16 conversions (weights + query) ----
    cvt_bf16_kernel<<<(A4 + 255) / 256, 256, 0, stream>>>(query, qbf, A4);
    cvt_bf16_kernel<<<(W4 + 255) / 256, 256, 0, stream>>>(Wq,  Wqb,  W4);
    cvt_bf16_kernel<<<(W4 + 255) / 256, 256, 0, stream>>>(Wv,  Wvb,  W4);
    cvt_bf16_kernel<<<(3 * W4 + 255) / 256, 256, 0, stream>>>(Wg1, Wg1b, 3 * W4);
    cvt_bf16_kernel<<<(W4 + 255) / 256, 256, 0, stream>>>(Wg2, Wg2b, W4);
    cvt_bf16_kernel<<<(W4 + 255) / 256, 256, 0, stream>>>(Wo,  Wob,  W4);
    transpose1024_bf<<<dim3(32, 32), dim3(32, 8), 0, stream>>>(Wk, WkTb);

    // q = query @ Wq^T (bf16 out only)
    wmma_gemm_async<1024, 0, 2><<<gemmGrid, 256, 0, stream>>>(
        qbf, qbf, qbf, Wqb, nullptr, nullptr, qpbf, nullptr, nullptr);
    // qk = q @ Wk (f32 out for softmax)
    wmma_gemm_async<1024, 0, 1><<<gemmGrid, 256, 0, stream>>>(
        qpbf, qpbf, qpbf, WkTb, nullptr, qk, nullptr, nullptr, nullptr);

    // streaming: scores -> softmax (attn out) -> attn-pooled memory (bf16)
    attn_pool<<<BH, 256, 0, stream>>>(qk, context_bank, memory_bank, attn, pooledbf);

    // retrieved = pooled @ Wv^T : f32 + bf16 + diffbf = bf16(query - retrieved)
    wmma_gemm_async<1024, 0, 7><<<gemmGrid, 256, 0, stream>>>(
        pooledbf, pooledbf, pooledbf, Wvb, nullptr, retrf, retrbf, query, diffbf);

    // g1 = relu([query,retr,query-retr] @ Wg1^T + bg1), K=3072 via region pointers
    wmma_gemm_async<3072, 2, 2><<<gemmGrid, 256, 0, stream>>>(
        qbf, retrbf, diffbf, Wg1b, bg1, nullptr, g1bf, nullptr, nullptr);

    // gate = sigmoid(g1 @ Wg2^T + bg2) (f32 out)
    wmma_gemm_async<1024, 3, 1><<<gemmGrid, 256, 0, stream>>>(
        g1bf, g1bf, g1bf, Wg2b, bg2, gate, nullptr, nullptr, nullptr);

    // fused = gate*query + (1-gate)*retr  (bf16 out)
    fuse_gate<<<(A4 + 255) / 256, 256, 0, stream>>>(gate, query, retrf, fusedbf, A4);

    // out = fused @ Wo^T + bo
    wmma_gemm_async<1024, 1, 1><<<gemmGrid, 256, 0, stream>>>(
        fusedbf, fusedbf, fusedbf, Wob, bo, out, nullptr, nullptr, nullptr);
}
